// AdaDCRN_VGAE_59639915872700
// MI455X (gfx1250) — matrix operand order repaired and checked by reading, not compiled, not run on hardware
//
#include <hip/hip_runtime.h>
#include <hip/hip_bf16.h>
#include <math.h>

// ---------------------------------------------------------------------------
// AdaDCRN / VGAE forward for MI455X (gfx1250, wave32, WMMA bf16 + async LDS)
// ---------------------------------------------------------------------------

typedef __bf16 bf16_t;
typedef __attribute__((ext_vector_type(16))) __bf16 bf16x16;
typedef __attribute__((ext_vector_type(8)))  float  f32x8;

#define NN   10000
#define INF  256
#define HIDF 256
#define ZF   64
#define NCF  10
#define EE   320000
#define MPAD 10048   // 157 * 64 : zero-padded row count for the SYRK operand

__device__ __forceinline__ bf16_t f2bf(float f) {
  unsigned u = __float_as_uint(f);
  unsigned r = u + 0x7FFFu + ((u >> 16) & 1u);   // round-to-nearest-even
  unsigned short h = (unsigned short)(r >> 16);
  bf16_t b;
  __builtin_memcpy(&b, &h, sizeof(b));
  return b;
}

__device__ __forceinline__ float wred_sum(float v) {
#pragma unroll
  for (int o = 16; o > 0; o >>= 1) v += __shfl_xor(v, o, 32);
  return v;
}
__device__ __forceinline__ float wred_max(float v) {
#pragma unroll
  for (int o = 16; o > 0; o >>= 1) v = fmaxf(v, __shfl_xor(v, o, 32));
  return v;
}

__device__ __forceinline__ void atomic_add_f32(float* p, float v) {
  __hip_atomic_fetch_add(p, v, __ATOMIC_RELAXED, __HIP_MEMORY_SCOPE_AGENT);
}

// async global->LDS 16B copy (ASYNCcnt tracked, no VGPR round trip)
__device__ __forceinline__ void async_cp16(const void* gsrc, void* lds_dst) {
  unsigned la = (unsigned)(uintptr_t)lds_dst;                 // low 32b = LDS offset
  unsigned long long ga = (unsigned long long)(uintptr_t)gsrc;
  asm volatile("global_load_async_to_lds_b128 %0, %1, off"
               :: "v"(la), "v"(ga) : "memory");
}
__device__ __forceinline__ void async_wait0() {
  asm volatile("s_wait_asynccnt 0x0" ::: "memory");
}

// ---------------------------------------------------------------------------
// Tiled WMMA GEMM: C[M,Ng] = act(A[M,K] @ B + bias)
//   transB==0 : B is [K,Ng] row-major ; transB==1 : B is [Ng,K] (A @ B^T)
//   preReluA  : relu A while converting to bf16 ; act: 0 none, 1 relu, 2 softplus
// Block = 128 threads (4 waves). Tile 64(M) x 64(N), K-step 32. K % 32 == 0.
// ---------------------------------------------------------------------------
__global__ __launch_bounds__(128)
void gemm_wmma_bf16(const float* __restrict__ A, const float* __restrict__ B,
                    float* __restrict__ C, const float* __restrict__ bias,
                    int M, int Ng, int K, int transB, int preReluA, int act) {
  __shared__ bf16_t As[64][32];   // [m][k]
  __shared__ bf16_t Bs[64][32];   // [n][k]

  const int tid  = threadIdx.x;
  const int lane = tid & 31;
  const int wv   = tid >> 5;
  const int m0   = blockIdx.y * 64;
  const int n0   = blockIdx.x * 64;
  const bool fullM = (m0 + 64 <= M);
  const bool fullN = (n0 + 64 <= Ng);

  f32x8 acc[4] = {};

  for (int k0 = 0; k0 < K; k0 += 32) {
    // ---- stage A tile (64x32) as bf16 ----
    if (fullM) {
      for (int i = tid; i < 64 * 32; i += 128) {
        int r = i >> 5, c = i & 31;
        float av = A[(size_t)(m0 + r) * K + (k0 + c)];
        if (preReluA) av = fmaxf(av, 0.0f);
        As[r][c] = f2bf(av);
      }
    } else {
      for (int i = tid; i < 64 * 32; i += 128) {
        int r = i >> 5, c = i & 31;
        int gm = m0 + r;
        float av = (gm < M) ? A[(size_t)gm * K + (k0 + c)] : 0.0f;
        if (preReluA) av = fmaxf(av, 0.0f);
        As[r][c] = f2bf(av);
      }
    }
    // ---- stage B tile (as [n][k]) ----
    if (transB) {
      for (int i = tid; i < 64 * 32; i += 128) {
        int r = i >> 5, c = i & 31;           // r = n, c = k
        int gn = n0 + r;
        float bv = (fullN || gn < Ng) ? B[(size_t)gn * K + (k0 + c)] : 0.0f;
        Bs[r][c] = f2bf(bv);
      }
    } else if (fullN) {
      for (int i = tid; i < 64 * 32; i += 128) {
        int kk = i >> 6, nn = i & 63;         // coalesced over nn
        Bs[nn][kk] = f2bf(B[(size_t)(k0 + kk) * Ng + (n0 + nn)]);
      }
    } else {
      for (int i = tid; i < 64 * 32; i += 128) {
        int kk = i >> 6, nn = i & 63;
        int gn = n0 + nn;
        float bv = (gn < Ng) ? B[(size_t)(k0 + kk) * Ng + gn] : 0.0f;
        Bs[nn][kk] = f2bf(bv);
      }
    }
    __syncthreads();

    // ---- A fragment (16x32, rows 16*wv .. 16*wv+15) ----
    const int mRow = (wv << 4) + (lane & 15);
    const int half = lane >> 4;
    bf16x16 af;
#pragma unroll
    for (int i = 0; i < 8; i++) {
      int kk = ((i < 4) ? (2 * i) : (2 * i + 8)) + 8 * half;
      af[2 * i]     = As[mRow][kk];
      af[2 * i + 1] = As[mRow][kk + 1];
    }

    // ---- 4 B fragments / wmma accumulate ----
#pragma unroll
    for (int t = 0; t < 4; t++) {
      const int colL = 16 * t + (lane & 15);
      bf16x16 bfr;
#pragma unroll
      for (int i = 0; i < 8; i++) {
        int kk = ((i < 4) ? (2 * i) : (2 * i + 8)) + 8 * half;
        bfr[2 * i]     = Bs[colL][kk];
        bfr[2 * i + 1] = Bs[colL][kk + 1];
      }
      acc[t] = __builtin_amdgcn_wmma_f32_16x16x32_bf16(
          false, af, false, bfr, (short)0, acc[t], false, false);
    }
    __syncthreads();
  }

  // ---- epilogue + store ----
  const int half = lane >> 4;
#pragma unroll
  for (int t = 0; t < 4; t++) {
    int gn = n0 + 16 * t + (lane & 15);
    if (gn >= Ng) continue;
    float bv = (bias != nullptr) ? bias[gn] : 0.0f;
#pragma unroll
    for (int r = 0; r < 8; r++) {
      int gm = m0 + 16 * wv + r + 8 * half;
      if (gm >= M) continue;
      float v = acc[t][r] + bv;
      if (act == 1) v = fmaxf(v, 0.0f);
      else if (act == 2) v = (v > 20.0f) ? v : log1pf(expf(v));
      C[(size_t)gm * Ng + gn] = v;
    }
  }
}

// ---------------------------------------------------------------------------
// Pack z (f32 [M,64]) into zero-padded bf16 [MPAD,64] for the SYRK kernel
// ---------------------------------------------------------------------------
__global__ __launch_bounds__(256)
void pack_bf16(const float* __restrict__ Z, bf16_t* __restrict__ Zb, int M) {
  int i = blockIdx.x * 256 + threadIdx.x;
  if (i >= MPAD * ZF) return;
  int rowi = i >> 6;
  float v = (rowi < M) ? Z[i] : 0.0f;
  Zb[i] = f2bf(v);
}

// ---------------------------------------------------------------------------
// adj_logits = z @ z^T  via WMMA, operand tiles staged with async global->LDS
// (store-bound: 400 MB -> nontemporal stores to keep L2 for the SPMM gathers)
// Zb is bf16 [MPAD,64], zero padded, so tile loads are never OOB.
// ---------------------------------------------------------------------------
__global__ __launch_bounds__(128)
void syrk_wmma_async(const bf16_t* __restrict__ Zb, float* __restrict__ C, int M) {
  __shared__ bf16_t As[64][64];
  __shared__ bf16_t Bs[64][64];

  const int tid  = threadIdx.x;
  const int lane = tid & 31;
  const int wv   = tid >> 5;
  const int m0   = blockIdx.y * 64;
  const int n0   = blockIdx.x * 64;

  // stage: 64 rows x 128B per operand, 16B async chunks (512 per operand)
#pragma unroll
  for (int j = 0; j < 4; j++) {
    int c = tid + 128 * j;          // 0..511
    int r = c >> 3;                 // row in tile
    int seg = (c & 7) * 8;          // bf16 element offset of 16B segment
    async_cp16(Zb + (size_t)(m0 + r) * ZF + seg, &As[r][seg]);
    async_cp16(Zb + (size_t)(n0 + r) * ZF + seg, &Bs[r][seg]);
  }
  async_wait0();
  __syncthreads();

  f32x8 acc[4] = {};
  const int mRow = (wv << 4) + (lane & 15);
  const int half = lane >> 4;

#pragma unroll
  for (int k0 = 0; k0 < ZF; k0 += 32) {
    bf16x16 af;
#pragma unroll
    for (int i = 0; i < 8; i++) {
      int kk = k0 + ((i < 4) ? (2 * i) : (2 * i + 8)) + 8 * half;
      af[2 * i]     = As[mRow][kk];
      af[2 * i + 1] = As[mRow][kk + 1];
    }
#pragma unroll
    for (int t = 0; t < 4; t++) {
      const int colL = 16 * t + (lane & 15);
      bf16x16 bfr;
#pragma unroll
      for (int i = 0; i < 8; i++) {
        int kk = k0 + ((i < 4) ? (2 * i) : (2 * i + 8)) + 8 * half;
        bfr[2 * i]     = Bs[colL][kk];
        bfr[2 * i + 1] = Bs[colL][kk + 1];
      }
      acc[t] = __builtin_amdgcn_wmma_f32_16x16x32_bf16(
          false, af, false, bfr, (short)0, acc[t], false, false);
    }
  }

#pragma unroll
  for (int t = 0; t < 4; t++) {
    int gn = n0 + 16 * t + (lane & 15);
    if (gn >= M) continue;
#pragma unroll
    for (int r = 0; r < 8; r++) {
      int gm = m0 + 16 * wv + r + 8 * half;
      if (gm >= M) continue;
      __builtin_nontemporal_store(acc[t][r], &C[(size_t)gm * M + gn]);
    }
  }
}

// ---------------------------------------------------------------------------
// SPMM scatter: Y[row[e],:] += vals[e] * X[col[e],:]   (D = 256, float4 lanes)
// ---------------------------------------------------------------------------
__global__ __launch_bounds__(256)
void spmm_scatter(const int* __restrict__ row, const int* __restrict__ col,
                  const float* __restrict__ vals, const float* __restrict__ X,
                  float* __restrict__ Y, int E) {
  long long tid = (long long)blockIdx.x * blockDim.x + threadIdx.x;
  if (tid >= (long long)E * 64) return;
  int e = (int)(tid >> 6);
  int q = (int)(tid & 63);
  int c = col[e], r = row[e];
  float v = vals[e];
  const float4 xv = ((const float4*)(X + (size_t)c * HIDF))[q];
  float* y = Y + (size_t)r * HIDF + q * 4;
  atomic_add_f32(y + 0, xv.x * v);
  atomic_add_f32(y + 1, xv.y * v);
  atomic_add_f32(y + 2, xv.z * v);
  atomic_add_f32(y + 3, xv.w * v);
}

// ---------------------------------------------------------------------------
// Per-node attention dots: a[n]=h_nb[n]·attW[0:256], b[n]=h_sf[n]·attW[256:512]
// ---------------------------------------------------------------------------
__global__ __launch_bounds__(256)
void att_dot(const float* __restrict__ hnb, const float* __restrict__ hsf,
             const float* __restrict__ attW, float* __restrict__ aN,
             float* __restrict__ bN, int n) {
  int wv = threadIdx.x >> 5, lane = threadIdx.x & 31;
  int node = blockIdx.x * 8 + wv;
  if (node >= n) return;
  float sa = 0.f, sb = 0.f;
  for (int j = lane; j < HIDF; j += 32) {
    sa += hnb[(size_t)node * HIDF + j] * attW[j];
    sb += hsf[(size_t)node * HIDF + j] * attW[HIDF + j];
  }
  sa = wred_sum(sa);
  sb = wred_sum(sb);
  if (lane == 0) { aN[node] = sa; bN[node] = sb; }
}

// ---------------------------------------------------------------------------
// Edge gate: hard-concrete mask, rowsum scatter, l0 accumulation
// ---------------------------------------------------------------------------
__global__ __launch_bounds__(256)
void edge_gate(const int* __restrict__ row, const int* __restrict__ col,
               const float* __restrict__ vals, const float* __restrict__ aN,
               const float* __restrict__ bN, const float* __restrict__ attb,
               float* __restrict__ mv, float* __restrict__ rowsum,
               float* __restrict__ l0, int E) {
  int e = blockIdx.x * 256 + threadIdx.x;
  float term = 0.0f;
  if (e < E) {
    float w = aN[row[e]] + bN[col[e]] + attb[0];
    float gate = 1.0f / (1.0f + expf(-w));
    float mask = fminf(fmaxf(gate * 1.6f - 0.5f, 0.0f), 1.0f); // (ZETA-GAMMA), GAMMA
    float m = vals[e] * mask;
    mv[e] = m;
    atomic_add_f32(&rowsum[row[e]], m);
    float lw = w + 0.78845736f;                 // w - log(-GAMMA/ZETA)
    term = (1.0f / (1.0f + expf(-lw))) * (1.0f / (float)EE);
  }
  term = wred_sum(term);
  if ((threadIdx.x & 31) == 0 && term != 0.0f) atomic_add_f32(l0, term);
}

__global__ __launch_bounds__(256)
void node_deg(const float* __restrict__ rowsum, float* __restrict__ dN, int n) {
  int i = blockIdx.x * 256 + threadIdx.x;
  if (i >= n) return;
  float r = rowsum[i] + 1e-10f;
  dN[i] = fminf(fmaxf(rsqrtf(r), 0.0f), 10.0f);
}

__global__ __launch_bounds__(256)
void edge_norm(const int* __restrict__ row, const int* __restrict__ col,
               const float* __restrict__ mv, const float* __restrict__ dN,
               float* __restrict__ nv, int E) {
  int e = blockIdx.x * 256 + threadIdx.x;
  if (e >= E) return;
  nv[e] = mv[e] * dN[row[e]] * dN[col[e]];
}

// ---------------------------------------------------------------------------
// Attention fusion: scores via tanh MLP, softmax over 2 views (TEMP=0.5)
// ---------------------------------------------------------------------------
__global__ __launch_bounds__(256)
void fusion(const float* __restrict__ zg, const float* __restrict__ zd,
            const float* __restrict__ f1W, const float* __restrict__ f1b,
            const float* __restrict__ f2W, float* __restrict__ zf,
            float* __restrict__ wt, int n) {
  int wv = threadIdx.x >> 5, lane = threadIdx.x & 31;
  int node = blockIdx.x * 8 + wv;
  if (node >= n) return;
  const float* g = zg + (size_t)node * ZF;
  const float* d = zd + (size_t)node * ZF;
  float sg = f1b[lane], sd = f1b[lane];
  for (int k = 0; k < ZF; k++) {
    float w = f1W[k * 32 + lane];
    sg += g[k] * w;
    sd += d[k] * w;
  }
  float f2 = f2W[lane];
  float scoreg = wred_sum(tanhf(sg) * f2);
  float scored = wred_sum(tanhf(sd) * f2);
  float l0 = scoreg * 2.0f, l1 = scored * 2.0f;     // /TEMP
  float m = fmaxf(l0, l1);
  float e0 = expf(l0 - m), e1 = expf(l1 - m);
  float inv = 1.0f / (e0 + e1);
  float w0 = e0 * inv, w1 = e1 * inv;
  for (int k = lane; k < ZF; k += 32)
    zf[(size_t)node * ZF + k] = w0 * g[k] + w1 * d[k];
  if (lane == 0) { wt[node * 2] = w0; wt[node * 2 + 1] = w1; }
}

// ---------------------------------------------------------------------------
// Cluster head: q = softmax(z @ head_W + head_b) over 10 classes
// ---------------------------------------------------------------------------
__global__ __launch_bounds__(256)
void head_softmax(const float* __restrict__ Zm, const float* __restrict__ W,
                  const float* __restrict__ b, float* __restrict__ Q, int n) {
  int wv = threadIdx.x >> 5, lane = threadIdx.x & 31;
  int node = blockIdx.x * 8 + wv;
  if (node >= n) return;
  float logit = -1e30f;
  if (lane < NCF) {
    float s = b[lane];
    for (int j = 0; j < ZF; j++) s += Zm[(size_t)node * ZF + j] * W[j * NCF + lane];
    logit = s;
  }
  float mx = wred_max(logit);
  float e = (lane < NCF) ? expf(logit - mx) : 0.0f;
  float se = wred_sum(e);
  if (lane < NCF) Q[(size_t)node * NCF + lane] = e / se;
}

// ---------------------------------------------------------------------------
extern "C" void kernel_launch(void* const* d_in, const int* in_sizes, int n_in,
                              void* d_out, int out_size, void* d_ws, size_t ws_size,
                              hipStream_t stream) {
  (void)in_sizes; (void)n_in; (void)out_size; (void)ws_size;

  const float* x        = (const float*)d_in[0];
  const int*   row      = (const int*)d_in[1];
  const int*   col      = (const int*)d_in[2];
  const float* adj_vals = (const float*)d_in[3];
  const float* gcn_W = (const float*)d_in[4];  const float* gcn_b = (const float*)d_in[5];
  const float* m1_W  = (const float*)d_in[6];  const float* m1_b  = (const float*)d_in[7];
  const float* m2_W  = (const float*)d_in[8];  const float* m2_b  = (const float*)d_in[9];
  const float* s1_W  = (const float*)d_in[10]; const float* s1_b  = (const float*)d_in[11];
  const float* s2_W  = (const float*)d_in[12]; const float* s2_b  = (const float*)d_in[13];
  const float* nb_W  = (const float*)d_in[14]; const float* nb_b  = (const float*)d_in[15];
  const float* sf_W  = (const float*)d_in[16]; const float* sf_b  = (const float*)d_in[17];
  const float* att_W = (const float*)d_in[18]; const float* att_b = (const float*)d_in[19];
  const float* f1_W  = (const float*)d_in[20]; const float* f1_b  = (const float*)d_in[21];
  const float* f2_W  = (const float*)d_in[22];
  const float* head_W = (const float*)d_in[23]; const float* head_b = (const float*)d_in[24];
  const float* d1_W  = (const float*)d_in[25]; const float* d1_b  = (const float*)d_in[26];
  const float* d2_W  = (const float*)d_in[27]; const float* d2_b  = (const float*)d_in[28];

  // ---- output layout (concatenated, fp32) ----
  float* out      = (float*)d_out;
  float* q_fused  = out;                                  // N*NC
  float* q_gen    = q_fused + (size_t)NN * NCF;
  float* q_den    = q_gen + (size_t)NN * NCF;
  float* adj      = q_den + (size_t)NN * NCF;             // N*N
  float* zf       = adj + (size_t)NN * NN;                // N*Z
  float* mu       = zf + (size_t)NN * ZF;                 // N*Z  (== z_gen)
  float* logstd   = mu + (size_t)NN * ZF;                 // N*Z
  float* l0       = logstd + (size_t)NN * ZF;             // 1
  float* zden     = l0 + 1;                               // N*Z
  float* wts      = zden + (size_t)NN * ZF;               // N*2
  float* recon    = wts + (size_t)NN * 2;                 // N*IN

  // ---- workspace layout ----
  float* ws   = (float*)d_ws;
  float* xW   = ws;                                  // N*HID
  float* buf1 = xW + (size_t)NN * HIDF;              // N*HID (hidden / h_nb / hidden2)
  float* buf2 = buf1 + (size_t)NN * HIDF;            // N*HID (h_sf / recon-hidden)
  float* t    = buf2 + (size_t)NN * HIDF;            // N*Z
  float* mv   = t + (size_t)NN * ZF;                 // E
  float* nv   = mv + (size_t)EE;                     // E
  float* aN   = nv + (size_t)EE;                     // N
  float* bN   = aN + NN;                             // N
  float* rs   = bN + NN;                             // N
  float* dN   = rs + NN;                             // N
  bf16_t* zbf = (bf16_t*)(dN + NN);                  // MPAD*Z bf16 (padded)

  auto gemm = [&](const float* A, const float* B, float* C, const float* bias,
                  int M, int Ng, int K, int transB, int preReluA, int act) {
    dim3 g((unsigned)((Ng + 63) / 64), (unsigned)((M + 63) / 64));
    gemm_wmma_bf16<<<g, 128, 0, stream>>>(A, B, C, bias, M, Ng, K, transB, preReluA, act);
  };

  const int spmmBlocks = (int)(((long long)EE * 64 + 255) / 256);
  const int nodeWaveBlocks = (NN + 7) / 8;

  // 1) xW = x @ gcn_W + gcn_b     (shared between both encodes)
  gemm(x, gcn_W, xW, gcn_b, NN, HIDF, INF, 0, 0, 0);

  // 2) generative encode: hidden = spmm(adj_vals, xW)  (relu fused downstream)
  hipMemsetAsync(buf1, 0, (size_t)NN * HIDF * sizeof(float), stream);
  spmm_scatter<<<spmmBlocks, 256, 0, stream>>>(row, col, adj_vals, xW, buf1, EE);
  gemm(buf1, m1_W, t, m1_b, NN, ZF, HIDF, 0, /*preReluA*/1, /*relu*/1);
  gemm(t, m2_W, mu, m2_b, NN, ZF, ZF, 0, 0, 0);                 // z_gen == mu
  gemm(buf1, s1_W, t, s1_b, NN, ZF, HIDF, 0, 1, 1);
  gemm(t, s2_W, logstd, s2_b, NN, ZF, ZF, 0, 0, /*softplus*/2);

  // 3) denoiser: per-node factorization of the edge MLP
  gemm(x, nb_W, buf1, nb_b, NN, HIDF, INF, 0, 0, 1);            // h_nb
  gemm(x, sf_W, buf2, sf_b, NN, HIDF, INF, 0, 0, 1);            // h_sf
  att_dot<<<nodeWaveBlocks, 256, 0, stream>>>(buf1, buf2, att_W, aN, bN, NN);
  hipMemsetAsync(rs, 0, (size_t)NN * sizeof(float), stream);
  hipMemsetAsync(l0, 0, sizeof(float), stream);
  edge_gate<<<(EE + 255) / 256, 256, 0, stream>>>(row, col, adj_vals, aN, bN, att_b,
                                                  mv, rs, l0, EE);
  node_deg<<<(NN + 255) / 256, 256, 0, stream>>>(rs, dN, NN);
  edge_norm<<<(EE + 255) / 256, 256, 0, stream>>>(row, col, mv, dN, nv, EE);

  // 4) denoised encode (reuses xW)
  hipMemsetAsync(buf1, 0, (size_t)NN * HIDF * sizeof(float), stream);
  spmm_scatter<<<spmmBlocks, 256, 0, stream>>>(row, col, nv, xW, buf1, EE);
  gemm(buf1, m1_W, t, m1_b, NN, ZF, HIDF, 0, 1, 1);
  gemm(t, m2_W, zden, m2_b, NN, ZF, ZF, 0, 0, 0);               // z_den

  // 5) adj_logits = z_gen @ z_gen^T  (async LDS staging + NT stores)
  pack_bf16<<<(MPAD * ZF + 255) / 256, 256, 0, stream>>>(mu, zbf, NN);
  {
    dim3 ga((unsigned)((NN + 63) / 64), (unsigned)((NN + 63) / 64));
    syrk_wmma_async<<<ga, 128, 0, stream>>>(zbf, adj, NN);
  }

  // 6) attention fusion + heads
  fusion<<<nodeWaveBlocks, 256, 0, stream>>>(mu, zden, f1_W, f1_b, f2_W, zf, wts, NN);
  head_softmax<<<nodeWaveBlocks, 256, 0, stream>>>(zf,   head_W, head_b, q_fused, NN);
  head_softmax<<<nodeWaveBlocks, 256, 0, stream>>>(mu,   head_W, head_b, q_gen, NN);
  head_softmax<<<nodeWaveBlocks, 256, 0, stream>>>(zden, head_W, head_b, q_den, NN);

  // 7) decoder: recon_den = relu(z_den @ d1_W + b) @ d2_W + b
  gemm(zden, d1_W, buf2, d1_b, NN, HIDF, ZF, 0, 0, 1);
  gemm(buf2, d2_W, recon, d2_b, NN, INF, HIDF, 0, 0, 0);
}
